// ProbAttention_45079976739601
// MI455X (gfx1250) — compile-verified, matrix operand order checked by reading, not run on hardware
//
#include <hip/hip_runtime.h>
#include <cmath>
#include <cstdint>

// ProbAttention (Informer ProbSparse) for MI455X / gfx1250, faithful to the
// reference's quirks (scores = Q_reduce, softmax over D, scatter by int(K[...,0,:])).
//
// Pipeline (all on `stream`):
//   K0: zero d_out (134MB) with b128 stores          -> bandwidth
//   K1: M[b,h,l] = max_s q.k_s - sum_s q.k_s / LK    -> TDM async Q-tile load to LDS,
//       gathered K rows via global b128 (L2-resident), wave32 shuffle reductions
//   K2: per (b,h): top-45 of M, row softmax of Q_reduce, attn @ V[:64] via
//       v_wmma_f32_16x16x4_f32 (f32-precision WMMA), ordered scatter.

namespace {

constexpr int kB  = 8;
constexpr int kLQ = 4096;
constexpr int kLK = 4096;
constexpr int kH  = 16;
constexpr int kD  = 64;
constexpr int kQPB = 64;   // queries per block in the M kernel

typedef float        v2f  __attribute__((ext_vector_type(2)));
typedef float        v8f  __attribute__((ext_vector_type(8)));
typedef _Float16     v16h __attribute__((ext_vector_type(16)));
typedef unsigned int u32x4 __attribute__((ext_vector_type(4)));
typedef int          i32x4 __attribute__((ext_vector_type(4)));
typedef int          i32x8 __attribute__((ext_vector_type(8)));

#if __has_builtin(__builtin_amdgcn_tensor_load_to_lds) && __has_builtin(__builtin_amdgcn_s_wait_tensorcnt)
#define PA_HAVE_TDM 1
#endif

} // namespace

// ---------------------------------------------------------------- K0: zero out
__global__ __launch_bounds__(256) void pa_zero_kernel(float4* __restrict__ p, long n4) {
    long i = (long)blockIdx.x * blockDim.x + threadIdx.x;
    long stride = (long)gridDim.x * blockDim.x;
    float4 z = make_float4(0.f, 0.f, 0.f, 0.f);
    for (; i < n4; i += stride) p[i] = z;
}

// ----------------------------------------------------- K1: sampled-QK metric M
// grid: (LQ/kQPB, H, B), block: 256 (8 waves of 32)
__global__ __launch_bounds__(256) void pa_m_kernel(const float* __restrict__ Q,
                                                   const float* __restrict__ K,
                                                   const int*   __restrict__ samp,
                                                   float*       __restrict__ Mout,
                                                   int sample_k) {
    __shared__ float Qtile[kQPB * kD];   // 16 KB

    const int b  = blockIdx.z;
    const int h  = blockIdx.y;
    const int q0 = blockIdx.x * kQPB;
    const size_t qbase = (((size_t)b * kLQ + q0) * kH + h) * kD;

#ifdef PA_HAVE_TDM
    // Async Tensor Data Mover: 2D tile (kQPB rows x kD f32), row stride H*D elems,
    // packed contiguously into LDS.  Issued by wave 0 only.
    if (threadIdx.x < 32) {
        unsigned lds_off = (unsigned)(uintptr_t)(void*)&Qtile[0];
        unsigned long long ga = (unsigned long long)(uintptr_t)(Q + qbase);
        u32x4 g0 = {};
        g0[0] = 1u;                                           // count=1 valid descriptor
        g0[1] = lds_off;                                      // lds_addr
        g0[2] = (unsigned)(ga & 0xFFFFFFFFu);                 // global_addr[31:0]
        g0[3] = (unsigned)((ga >> 32) & 0x01FFFFFFu) | (2u << 30); // addr[56:32] | type=2
        unsigned td1 = (unsigned)(kLQ - q0);                  // rows remaining (OOB bound)
        i32x8 g1 = {};
        g1[0] = (int)(2u << 16);                              // data_size=2 (4 bytes)
        g1[1] = (int)(((unsigned)kD & 0xFFFFu) << 16);        // tensor_dim0[15:0]
        g1[2] = (int)(((td1 & 0xFFFFu) << 16));               // dim0 hi=0 | tensor_dim1 lo
        g1[3] = (int)((((unsigned)kD & 0xFFFFu) << 16) | ((td1 >> 16) & 0xFFFFu)); // tile_dim0|dim1 hi
        g1[4] = (int)((unsigned)kQPB);                        // tile_dim1, tile_dim2=0
        g1[5] = (int)(kH * kD);                               // tensor_dim0_stride[31:0]
        g1[6] = 0;
        g1[7] = 0;
        i32x4 gz = {0, 0, 0, 0};
#if __has_include(<hip/amd_detail/amd_gfx1250_TDM.h>)
        i32x8 z8 = {0, 0, 0, 0, 0, 0, 0, 0};
        __builtin_amdgcn_tensor_load_to_lds(g0, g1, gz, gz, z8, 0);
#else
        __builtin_amdgcn_tensor_load_to_lds(g0, g1, gz, gz, 0);
#endif
        __builtin_amdgcn_s_wait_tensorcnt(0);
    }
    __syncthreads();
#else
    for (int i = threadIdx.x; i < kQPB * (kD / 4); i += 256) {
        int r = i >> 4, c4 = i & 15;
        ((float4*)Qtile)[i] = *(const float4*)(Q + qbase + (size_t)r * kH * kD + c4 * 4);
    }
    __syncthreads();
#endif

    const int wave = threadIdx.x >> 5;
    const int lane = threadIdx.x & 31;

    for (int qi = wave; qi < kQPB; qi += 8) {
        const int l = q0 + qi;
        const float4* __restrict__ qrow = (const float4*)&Qtile[qi * kD];
        float lmax = -__builtin_inff();
        float lsum = 0.f;
        for (int s = lane; s < sample_k; s += 32) {
            const int ki = samp[(size_t)l * sample_k + s];
            const float4* __restrict__ krow =
                (const float4*)(K + (((size_t)b * kLK + ki) * kH + h) * kD);
            float acc = 0.f;
#pragma unroll
            for (int j = 0; j < kD / 4; ++j) {
                float4 a = qrow[j];
                float4 c = krow[j];
                acc += a.x * c.x + a.y * c.y + a.z * c.z + a.w * c.w;
            }
            lmax = fmaxf(lmax, acc);
            lsum += acc;
        }
        // wave32 cross-lane reduction
        for (int off = 16; off > 0; off >>= 1) {
            lmax = fmaxf(lmax, __shfl_xor(lmax, off, 32));
            lsum += __shfl_xor(lsum, off, 32);
        }
        if (lane == 0)
            Mout[((size_t)b * kH + h) * kLQ + l] = lmax - lsum * (1.0f / (float)kLK);
    }
}

// ------------------------------- K2: top-k + softmax + WMMA GEMM + scatter
// grid: (H, B), block 256 (8 waves)
__global__ __launch_bounds__(256) void pa_ctx_kernel(const float* __restrict__ Q,
                                                     const float* __restrict__ K,
                                                     const float* __restrict__ V,
                                                     const float* __restrict__ Mws,
                                                     float*       __restrict__ out,
                                                     int u) {
    __shared__ float Msh[kLQ];        // 16 KB
    __shared__ float bestV[256];      //  1 KB
    __shared__ int   bestI[256];      //  1 KB
    __shared__ int   Mtop[64];
    __shared__ float P[48][kD];       // 12 KB   softmax(Q_reduce), zero-padded to 48 rows
    __shared__ float Vs[kD][kD];      // 16 KB   V[:, :, :64, :]
    __shared__ float ctx[kD][kD];     // 16 KB   ctx_upd padded to 64 rows
    __shared__ int   idxsh[kD];

    const int h = blockIdx.x;
    const int b = blockIdx.y;
    const int tid  = threadIdx.x;
    const int wave = tid >> 5;
    const int lane = tid & 31;
    const int mtiles = (u + 15) >> 4;           // 3 for u=45
    const int mrows  = mtiles * 16;             // 48

    // --- load M for this (b,h)
    const float* __restrict__ Mrow = Mws + ((size_t)b * kH + h) * kLQ;
    for (int i = tid; i < kLQ; i += 256) Msh[i] = Mrow[i];
    __syncthreads();

    // --- iterative top-k (descending; ties -> lowest index, matching lax.top_k)
    for (int k = 0; k < u; ++k) {
        float bv = -__builtin_inff();
        int   bi = 0x7FFFFFFF;
        for (int i = tid; i < kLQ; i += 256) {
            float v = Msh[i];
            if (v > bv) { bv = v; bi = i; }
        }
        bestV[tid] = bv; bestI[tid] = bi;
        __syncthreads();
        for (int s = 128; s > 0; s >>= 1) {
            if (tid < s) {
                float ov = bestV[tid + s]; int oi = bestI[tid + s];
                if (ov > bestV[tid] || (ov == bestV[tid] && oi < bestI[tid])) {
                    bestV[tid] = ov; bestI[tid] = oi;
                }
            }
            __syncthreads();
        }
        if (tid == 0) { Mtop[k] = bestI[0]; Msh[bestI[0]] = -__builtin_inff(); }
        __syncthreads();
    }

    // --- load V[:64,:64] tile
    for (int i = tid; i < kD * (kD / 4); i += 256) {
        int r = i >> 4, c4 = i & 15;
        ((float4*)&Vs[r][0])[c4] =
            *(const float4*)(V + (((size_t)b * kLK + r) * kH + h) * kD + c4 * 4);
    }

    // --- row softmax of Q_reduce into P (rows [u, mrows) zero-padded)
    for (int r = wave; r < mrows; r += 8) {
        if (r < u) {
            const float* __restrict__ qrow = Q + (((size_t)b * kLQ + Mtop[r]) * kH + h) * kD;
            float e0 = qrow[lane * 2], e1 = qrow[lane * 2 + 1];
            float m = fmaxf(e0, e1);
            for (int off = 16; off > 0; off >>= 1) m = fmaxf(m, __shfl_xor(m, off, 32));
            float p0 = __expf(e0 - m), p1 = __expf(e1 - m);
            float s = p0 + p1;
            for (int off = 16; off > 0; off >>= 1) s += __shfl_xor(s, off, 32);
            float inv = 1.0f / s;
            P[r][lane * 2]     = p0 * inv;
            P[r][lane * 2 + 1] = p1 * inv;
        } else {
            P[r][lane * 2] = 0.f;
            P[r][lane * 2 + 1] = 0.f;
        }
    }
    // ctx rows beyond the GEMM output are the reference's zero padding
    for (int i = tid; i < (kD - mrows) * kD; i += 256)
        ctx[mrows + i / kD][i % kD] = 0.f;
    __syncthreads();

    // --- GEMM: ctx[0:mrows, 0:64] = P @ Vs via WMMA (tiles of 16x16, K stepped)
    const int ntiles = mtiles * 4;
    for (int t = wave; t < ntiles; t += 8) {
        const int mt = t >> 2, nt = t & 3;
        const int m  = mt * 16 + (lane & 15);
        const int n  = nt * 16 + (lane & 15);
        v8f acc = {};
#if __has_builtin(__builtin_amdgcn_wmma_f32_16x16x4_f32)
        // f32 A 16x4: lanes0-15 hold K=k+0/k+1, lanes16-31 K=k+2/k+3 (per ISA layout);
        // f32 B 4x16 mirrors C/D striping: VGPR v -> K=v (+2 for upper lanes).
        const int klo = (lane < 16) ? 0 : 2;
        for (int k4 = 0; k4 < kD; k4 += 4) {
            v2f a, bb;
            a.x  = P[m][k4 + klo];
            a.y  = P[m][k4 + klo + 1];
            bb.x = Vs[k4 + klo][n];
            bb.y = Vs[k4 + klo + 1][n];
            acc = __builtin_amdgcn_wmma_f32_16x16x4_f32(false, a, false, bb,
                                                        (short)0, acc, false, false);
        }
#else
        // fallback: codegen-confirmed f16 16x16x32 WMMA with f32 accumulate
        const int half = (lane < 16) ? 0 : 8;
        for (int kb = 0; kb < kD; kb += 32) {
            v16h a, bm;
#pragma unroll
            for (int e = 0; e < 16; ++e) {
                int v = e >> 1, pos = e & 1;
                int kk = ((v < 4) ? (2 * v) : (16 + 2 * (v - 4))) + half + pos;
                a[e]  = (_Float16)P[m][kb + kk];
                bm[e] = (_Float16)Vs[kb + kk][n];
            }
            acc = __builtin_amdgcn_wmma_f32_16x16x32_f16(false, a, false, bm,
                                                         (short)0, acc, false, false);
        }
#endif
        // C/D layout: VGPR r -> row mt*16 + r (+8 for upper lanes), col nt*16 + (lane&15)
        const int rbase = mt * 16 + ((lane < 16) ? 0 : 8);
#pragma unroll
        for (int r = 0; r < 8; ++r) ctx[rbase + r][nt * 16 + (lane & 15)] = acc[r];
    }

    // --- scatter indices: clip(int(K[b,h,0,:]), 0, LQ-1)
    if (tid < kD) {
        float kv = K[(((size_t)b * kLK) * kH + h) * kD + tid];
        int iv = (int)kv;                       // trunc toward zero == astype(int32)
        idxsh[tid] = iv < 0 ? 0 : (iv > kLQ - 1 ? kLQ - 1 : iv);
    }
    __syncthreads();

    // --- ordered scatter by a single wave (deterministic for duplicate indices)
    if (wave == 0) {
        for (int d = 0; d < kD; ++d) {
            const int row = idxsh[d];
            float2 val;
            val.x = ctx[d][lane * 2];
            val.y = ctx[d][lane * 2 + 1];
            *(float2*)(out + (((size_t)b * kLQ + row) * kH + h) * kD + lane * 2) = val;
        }
    }
}

// ------------------------------------------------------------------- launcher
extern "C" void kernel_launch(void* const* d_in, const int* in_sizes, int n_in,
                              void* d_out, int out_size, void* d_ws, size_t ws_size,
                              hipStream_t stream) {
    const float* Q    = (const float*)d_in[0];
    const float* K    = (const float*)d_in[1];
    const float* V    = (const float*)d_in[2];
    const int*   samp = (const int*)d_in[3];
    float* out = (float*)d_out;
    float* Mws = (float*)d_ws;   // B*H*LQ floats = 2 MB scratch

    const int sample_k = in_sizes[3] / kLQ;                       // 45
    int u = (int)fmin(5.0 * ceil(log((double)kLQ)), (double)kLQ); // 45
    if (u > 48) u = 48;                                           // LDS tiling bound

    const long n4 = (long)kB * kLQ * kH * kD / 4;
    pa_zero_kernel<<<4096, 256, 0, stream>>>((float4*)out, n4);

    pa_m_kernel<<<dim3(kLQ / kQPB, kH, kB), 256, 0, stream>>>(Q, K, samp, Mws, sample_k);

    pa_ctx_kernel<<<dim3(kH, kB), 256, 0, stream>>>(Q, K, V, Mws, out, u);
}
